// decoderLSTM_43834436223394
// MI455X (gfx1250) — matrix operand order repaired and checked by reading, not compiled, not run on hardware
//
#include <hip/hip_runtime.h>

// ---------------------------------------------------------------------------
// LSTM decoder (N=128, H=1024, D=1024, T=256) as a persistent WMMA kernel.
//   step 0  : gates = hT @ W_ih^T + (b_ih+b_hh)
//   step s>0: gates = h_{s-1} @ (W_ih+W_hh)^T + (b_ih+b_hh)
//   out[:,s,:] = h_s @ W_fc^T + b_fc
//
// 64 WGs x 256 thr = 512 wave32. Wave (blockIdx=jt, waveIdx=mt) owns one
// 16x16 tile quad (i,f,g,o) + one FC tile, full K=1024 in registers.
// The per-WG weight panel (4 x W_comb tile-cols + 1 x W_fc tile-col, f16,
// 160 KB) is staged into LDS once and reused for all 256 steps; only the
// h matrix (L2-resident) travels the global path each step.
// Device-wide monotone-counter barrier between steps.
// ---------------------------------------------------------------------------

typedef _Float16 h16;
typedef __attribute__((ext_vector_type(16))) _Float16 v16h;
typedef __attribute__((ext_vector_type(8)))  _Float16 v8h;
typedef __attribute__((ext_vector_type(8)))  float    v8f;

#define N_B   128
#define HID   1024
#define GATES 4096
#define DOUT  1024
#define TSTEP 256
#define NWG   64
#define NTHR  256

#define LDS_TILES 160                 // 5 weight cols x 32 k-tiles
#define LDS_BYTES (LDS_TILES * 1024)  // 160 KB (fits 320 KB/WGP)

// ----------------------------- device helpers ------------------------------

__device__ inline float sigmoid_f(float x) { return 1.0f / (1.0f + __expf(-x)); }
__device__ inline float tanh_f(float x)    { return 1.0f - 2.0f / (1.0f + __expf(2.0f * x)); }

__device__ inline v8f vzero8() {
  v8f z = {0.f, 0.f, 0.f, 0.f, 0.f, 0.f, 0.f, 0.f};
  return z;
}

__device__ inline v8f wmma_f16(v16h a, v16h b, v8f c) {
  // D = A(16x32 f16) * B(32x16 f16) + C(16x16 f32)
  return __builtin_amdgcn_wmma_f32_16x16x32_f16(
      /*neg_a=*/false, a, /*neg_b=*/false, b,
      /*c_mod=*/(short)0, c, /*reuse_a=*/false, /*reuse_b=*/false);
}

union frag_u { v16h v; v8h h[2]; };

// A fragment from row-major global: arow pre-offset by row*HID + half*8.
__device__ inline v16h load_a_frag(const h16* __restrict__ arow, int ka) {
  frag_u u;
  u.h[0] = *(const v8h*)(arow + ka);
  u.h[1] = *(const v8h*)(arow + ka + 16);
  return u.v;
}

// B fragment from row-major global: brow pre-offset by col*HID + half*8.
__device__ inline v16h load_b_frag(const h16* __restrict__ brow, int ka) {
  frag_u u;
  u.h[0] = *(const v8h*)(brow + ka);
  u.h[1] = *(const v8h*)(brow + ka + 16);
  return u.v;
}

// B fragment from LDS panel. Tile layout: two 512B chunks of [lane][16B] --
// lanes 0-15 span all 64 banks per chunk access (conflict-free).
__device__ inline v16h load_b_lds(const h16* __restrict__ sm, int tile, int lane) {
  frag_u u;
  const h16* p = sm + tile * 512 + lane * 8;
  u.h[0] = *(const v8h*)(p);
  u.h[1] = *(const v8h*)(p + 256);
  return u.v;
}

// Device-wide barrier: monotone counter, target = NWG * phase.
__device__ inline void grid_barrier(unsigned* __restrict__ bar, unsigned target) {
  __threadfence();
  __syncthreads();
  if (threadIdx.x == 0) {
    __hip_atomic_fetch_add(bar, 1u, __ATOMIC_ACQ_REL, __HIP_MEMORY_SCOPE_AGENT);
    while (__hip_atomic_load(bar, __ATOMIC_ACQUIRE, __HIP_MEMORY_SCOPE_AGENT) < target) {
      __builtin_amdgcn_s_sleep(2);
    }
  }
  __syncthreads();
}

// ----------------------------- MMA inner loops -----------------------------

// Phase 0 only: four gate tiles from global W_ih.
__device__ inline void gates_mma_global(const h16* __restrict__ X, const h16* __restrict__ W,
                                        int mt, int colbase, int half, v8f acc[4]) {
  const h16* arow = X + (size_t)(mt * 16 + (colbase & 15)) * HID + half * 8;
  const h16* b0 = W + (size_t)colbase * HID + half * 8;
  const h16* b1 = b0 + (size_t)HID * HID;
  const h16* b2 = b1 + (size_t)HID * HID;
  const h16* b3 = b2 + (size_t)HID * HID;
#pragma unroll 4
  for (int ka = 0; ka < HID; ka += 32) {
    v16h a = load_a_frag(arow, ka);
    acc[0] = wmma_f16(a, load_b_frag(b0, ka), acc[0]);
    acc[1] = wmma_f16(a, load_b_frag(b1, ka), acc[1]);
    acc[2] = wmma_f16(a, load_b_frag(b2, ka), acc[2]);
    acc[3] = wmma_f16(a, load_b_frag(b3, ka), acc[3]);
  }
}

// Steady state: four gate tiles + FC tile, all B from LDS, one shared A frag.
__device__ inline void gates_fc_mma_lds(const h16* __restrict__ X, const h16* __restrict__ sm,
                                        int mt, int nl, int half, int lane,
                                        v8f acc[4], v8f& afc) {
  const h16* arow = X + (size_t)(mt * 16 + nl) * HID + half * 8;
#pragma unroll 4
  for (int kt = 0; kt < 32; ++kt) {
    v16h a = load_a_frag(arow, kt * 32);
    acc[0] = wmma_f16(a, load_b_lds(sm, 0 * 32 + kt, lane), acc[0]);
    acc[1] = wmma_f16(a, load_b_lds(sm, 1 * 32 + kt, lane), acc[1]);
    acc[2] = wmma_f16(a, load_b_lds(sm, 2 * 32 + kt, lane), acc[2]);
    acc[3] = wmma_f16(a, load_b_lds(sm, 3 * 32 + kt, lane), acc[3]);
    afc    = wmma_f16(a, load_b_lds(sm, 4 * 32 + kt, lane), afc);
  }
}

// ------------------------------- epilogues ---------------------------------

template <bool FIRST>
__device__ inline void lstm_epilogue(const v8f acc[4], const float* __restrict__ bias,
                                     float* __restrict__ cfrag, h16* __restrict__ hdst,
                                     int mt, int jt, int half, int nl) {
  const int cb = jt * 16 + nl;
  const float bi = bias[0 * HID + cb];
  const float bf = bias[1 * HID + cb];
  const float bg = bias[2 * HID + cb];
  const float bo = bias[3 * HID + cb];
#pragma unroll
  for (int v = 0; v < 8; ++v) {
    float iv = sigmoid_f(acc[0][v] + bi);
    float fv = sigmoid_f(acc[1][v] + bf);
    float gv = tanh_f(acc[2][v] + bg);
    float ov = sigmoid_f(acc[3][v] + bo);
    float cp = FIRST ? 0.0f : cfrag[v * 32];
    float cn = fv * cp + iv * gv;
    cfrag[v * 32] = cn;
    float hn = ov * tanh_f(cn);
    hdst[(size_t)(mt * 16 + v + half * 8) * HID + cb] = (h16)hn;
  }
}

__device__ inline void fc_store(const v8f acc, const float* __restrict__ bfc,
                                float* __restrict__ out, int step,
                                int mt, int jt, int half, int nl) {
  const int cb = jt * 16 + nl;
  const float bb = bfc[cb];
#pragma unroll
  for (int v = 0; v < 8; ++v) {
    const int row = mt * 16 + v + half * 8;
    out[((size_t)row * TSTEP + step) * DOUT + cb] = acc[v] + bb;
  }
}

// ------------------------------- kernels -----------------------------------

// Prologue: f32 -> f16 weight packing, combined weights/bias, barrier reset.
__global__ __launch_bounds__(256) void prep_kernel(
    const float* __restrict__ W_ih, const float* __restrict__ W_hh,
    const float* __restrict__ b_ih, const float* __restrict__ b_hh,
    const float* __restrict__ W_fc, const float* __restrict__ hT,
    h16* __restrict__ Wcomb, h16* __restrict__ Wih16, h16* __restrict__ Wfc16,
    h16* __restrict__ hT16, float* __restrict__ bias, unsigned* __restrict__ bar) {
  const size_t i = (size_t)blockIdx.x * blockDim.x + threadIdx.x;
  const size_t stride = (size_t)gridDim.x * blockDim.x;
  for (size_t k = i; k < (size_t)GATES * HID; k += stride) {
    float a = W_ih[k], b = W_hh[k];
    Wcomb[k] = (h16)(a + b);
    Wih16[k] = (h16)a;
  }
  for (size_t k = i; k < (size_t)DOUT * HID; k += stride) Wfc16[k] = (h16)W_fc[k];
  for (size_t k = i; k < (size_t)N_B * HID; k += stride) hT16[k] = (h16)hT[k];
  for (size_t k = i; k < (size_t)GATES; k += stride) bias[k] = b_ih[k] + b_hh[k];
  if (i == 0) *bar = 0u;
}

__global__ __launch_bounds__(NTHR, 1) void lstm_persistent(
    const h16* __restrict__ hT16,  // [128,1024]
    const h16* __restrict__ Wih16, // [4096,1024]  (step 0 only)
    const h16* __restrict__ Wcomb, // [4096,1024]  W_ih + W_hh
    const h16* __restrict__ Wfc16, // [1024,1024]
    const float* __restrict__ bias,// [4096] b_ih + b_hh
    const float* __restrict__ bfc, // [1024]
    h16* __restrict__ hbuf,        // [2][128][1024] double-buffered h
    float* __restrict__ cbuf,      // [128*1024] c-state in fragment layout
    float* __restrict__ out,       // [128,256,1024]
    unsigned* __restrict__ bar) {
  extern __shared__ h16 smem[];   // 160 KB weight panel, persistent all steps

  const int tid  = threadIdx.x;
  const int lane = tid & 31;
  const int mt   = tid >> 5;      // wave index in WG = batch tile 0..7
  const int jt   = blockIdx.x;    // hidden/output tile 0..63
  const int half = lane >> 4;     // 0: lanes 0-15, 1: lanes 16-31
  const int nl   = lane & 15;
  const int colbase = jt * 16 + nl;

  float* cfrag = cbuf + (size_t)((mt * 64 + jt) * 8) * 32 + lane;
  h16* h0 = hbuf;
  h16* h1 = hbuf + (size_t)N_B * HID;

  // ---- one-time LDS fill: 4 W_comb tile-cols + 1 W_fc tile-col, full K ----
  // Tile idx = g*32 + kt; each lane stores the exact 32B it will later read.
  for (int idx = mt; idx < LDS_TILES; idx += 8) {
    const int g  = idx >> 5;      // 0..3 gates, 4 = fc
    const int kt = idx & 31;
    const h16* W = (g < 4) ? (Wcomb + (size_t)g * HID * HID) : Wfc16;
    const h16* src = W + (size_t)colbase * HID + kt * 32 + half * 8;
    v8h c0 = *(const v8h*)(src);
    v8h c1 = *(const v8h*)(src + 16);
    h16* dst = smem + idx * 512 + lane * 8;
    *(v8h*)(dst)       = c0;
    *(v8h*)(dst + 256) = c1;
  }
  __syncthreads();

  // ---- phase 0: gates from hT via W_ih (global; one-time); cx = 0 ----
  {
    v8f acc[4] = {vzero8(), vzero8(), vzero8(), vzero8()};
    gates_mma_global(hT16, Wih16, mt, colbase, half, acc);
    lstm_epilogue<true>(acc, bias, cfrag, h0, mt, jt, half, nl);
  }
  grid_barrier(bar, (unsigned)NWG);

  // ---- phases 1..T-1: gates from h_{s-1} via LDS W_comb, + FC of h_{s-1} ----
  for (int s = 1; s < TSTEP; ++s) {
    const h16* hprev = (s & 1) ? h0 : h1;  // (s-1)&1
    h16* hnext = (s & 1) ? h1 : h0;        // s&1
    v8f acc[4] = {vzero8(), vzero8(), vzero8(), vzero8()};
    v8f afc = vzero8();
    gates_fc_mma_lds(hprev, smem, mt, nl, half, lane, acc, afc);
    lstm_epilogue<false>(acc, bias, cfrag, hnext, mt, jt, half, nl);
    fc_store(afc, bfc, out, s - 1, mt, jt, half, nl);
    grid_barrier(bar, (unsigned)NWG * (unsigned)(s + 1));
  }

  // ---- final: FC of h_{T-1} (B from LDS fc panel) ----
  {
    const h16* hlast = ((TSTEP - 1) & 1) ? h1 : h0;
    const h16* arow = hlast + (size_t)(mt * 16 + nl) * HID + half * 8;
    v8f afc = vzero8();
#pragma unroll 4
    for (int kt = 0; kt < 32; ++kt)
      afc = wmma_f16(load_a_frag(arow, kt * 32), load_b_lds(smem, 4 * 32 + kt, lane), afc);
    fc_store(afc, bfc, out, TSTEP - 1, mt, jt, half, nl);
  }
}

// ------------------------------- launcher ----------------------------------

extern "C" void kernel_launch(void* const* d_in, const int* in_sizes, int n_in,
                              void* d_out, int out_size, void* d_ws, size_t ws_size,
                              hipStream_t stream) {
  (void)in_sizes; (void)n_in; (void)out_size; (void)ws_size;
  const float* hT   = (const float*)d_in[0];
  const float* W_ih = (const float*)d_in[1];
  const float* W_hh = (const float*)d_in[2];
  const float* b_ih = (const float*)d_in[3];
  const float* b_hh = (const float*)d_in[4];
  const float* W_fc = (const float*)d_in[5];
  const float* b_fc = (const float*)d_in[6];
  float* out = (float*)d_out;

  // Workspace carve (all 256B aligned). Total ~20.2 MB.
  char* ws = (char*)d_ws;
  size_t off = 0;
  auto alloc = [&](size_t bytes) -> char* {
    char* p = ws + off;
    off += (bytes + 255) & ~(size_t)255;
    return p;
  };
  h16* Wcomb = (h16*)alloc((size_t)GATES * HID * sizeof(h16));
  h16* Wih16 = (h16*)alloc((size_t)GATES * HID * sizeof(h16));
  h16* Wfc16 = (h16*)alloc((size_t)DOUT * HID * sizeof(h16));
  h16* hT16  = (h16*)alloc((size_t)N_B * HID * sizeof(h16));
  h16* hbuf  = (h16*)alloc((size_t)2 * N_B * HID * sizeof(h16));
  float* cbuf = (float*)alloc((size_t)N_B * HID * sizeof(float));
  float* bias = (float*)alloc((size_t)GATES * sizeof(float));
  unsigned* bar = (unsigned*)alloc(256);

  hipLaunchKernelGGL(prep_kernel, dim3(1024), dim3(256), 0, stream,
                     W_ih, W_hh, b_ih, b_hh, W_fc, hT,
                     Wcomb, Wih16, Wfc16, hT16, bias, bar);
  hipLaunchKernelGGL(lstm_persistent, dim3(NWG), dim3(NTHR), LDS_BYTES, stream,
                     hT16, Wih16, Wcomb, Wfc16, bias, b_fc,
                     hbuf, cbuf, out, bar);
}
